// KernelAttention_16990890623407
// MI455X (gfx1250) — matrix operand order repaired and checked by reading, not compiled, not run on hardware
//
#include <hip/hip_runtime.h>

typedef __attribute__((ext_vector_type(16))) __bf16 v16bf;
typedef __attribute__((ext_vector_type(8)))  float  v8f;
typedef __attribute__((ext_vector_type(4)))  unsigned u32x4;
typedef __attribute__((ext_vector_type(8)))  unsigned u32x8;
typedef unsigned short u16;

// -------- helpers --------------------------------------------------------

__device__ __forceinline__ u16 f2bf(float f) {
  union { float f; unsigned u; } v; v.f = f;
  unsigned r = v.u + 0x7FFFu + ((v.u >> 16) & 1u);   // round-to-nearest-even
  return (u16)(r >> 16);
}

__device__ __forceinline__ v8f wmma_bf16(v16bf a, v16bf b, v8f c) {
  return __builtin_amdgcn_wmma_f32_16x16x32_bf16(
      /*neg_a=*/false, a, /*neg_b=*/false, b,
      /*c_mod=*/(short)0, c, /*reuse_a=*/false, /*reuse_b=*/false);
}

// A-matrix fragment (16x32 bf16, ISA 7.12.2): lane half h loads row dwords
// [dwbase+4h .. +3] and [dwbase+8+4h .. +3]  (two b128 loads)
__device__ __forceinline__ v16bf load_afrag(const u16* row, int dwbase, int half) {
  const uint4* p = (const uint4*)row;
  union { v16bf v; uint4 q[2]; } u;
  int i = dwbase >> 2;
  u.q[0] = p[i + half];
  u.q[1] = p[i + 2 + half];
  return u.v;
}

// B-matrix fragment (32x16 bf16): lane = column n, lane half kh loads row dwords
// [dwbase+4kh .. +3] and [dwbase+8+4kh .. +3] (two b128 loads, same K order as A).
__device__ __forceinline__ v16bf load_bfrag(const u16* row, int dwbase, int khalf) {
  const uint4* p = (const uint4*)row;
  union { v16bf v; uint4 q[2]; } u;
  int i = dwbase >> 2;
  u.q[0] = p[i + khalf];
  u.q[1] = p[i + 2 + khalf];
  return u.v;
}

// -------- fp32 -> bf16 convert ------------------------------------------

__global__ __launch_bounds__(256) void f32_to_bf16_kernel(
    const float* __restrict__ src, u16* __restrict__ dst, int n4) {
  int i = blockIdx.x * blockDim.x + threadIdx.x;
  if (i >= n4) return;
  float4 v = ((const float4*)src)[i];
  union { u16 s[4]; uint2 u; } o;
  o.s[0] = f2bf(v.x); o.s[1] = f2bf(v.y);
  o.s[2] = f2bf(v.z); o.s[3] = f2bf(v.w);
  ((uint2*)dst)[i] = o.u;
}

// -------- bf16 WMMA GEMM: out[M,N] = A[M,K] * W[N,K]^T (+bias) ----------
// One wave computes a 16x64 tile; 8 waves / 256-thread block.

__global__ __launch_bounds__(256) void gemm_bf16_wmma(
    const u16* __restrict__ A, const u16* __restrict__ W,
    u16* __restrict__ outb, float* __restrict__ outf,
    const float* __restrict__ bias, int M, int N, int K) {
  const int lane = threadIdx.x & 31;
  const int l15  = lane & 15;
  const int half = lane >> 4;
  const int nt64 = N >> 6;
  const int gid  = blockIdx.x * 8 + (threadIdx.x >> 5);
  const int mt   = gid / nt64;
  const int nt   = gid % nt64;
  if (mt * 16 >= M) return;

  const u16* arow = A + (size_t)(mt * 16 + l15) * K;
  const u16* brow0 = W + (size_t)(nt * 64 +  0 + l15) * K;
  const u16* brow1 = W + (size_t)(nt * 64 + 16 + l15) * K;
  const u16* brow2 = W + (size_t)(nt * 64 + 32 + l15) * K;
  const u16* brow3 = W + (size_t)(nt * 64 + 48 + l15) * K;

  v8f acc0 = {}, acc1 = {}, acc2 = {}, acc3 = {};
  const int ksteps = K >> 5;
  for (int kk = 0; kk < ksteps; ++kk) {
    v16bf a = load_afrag(arow, kk * 16, half);
    __builtin_prefetch(arow + (size_t)(kk + 1) * 32, 0, 0);
    acc0 = wmma_bf16(a, load_bfrag(brow0, kk * 16, half), acc0);
    acc1 = wmma_bf16(a, load_bfrag(brow1, kk * 16, half), acc1);
    acc2 = wmma_bf16(a, load_bfrag(brow2, kk * 16, half), acc2);
    acc3 = wmma_bf16(a, load_bfrag(brow3, kk * 16, half), acc3);
  }

  v8f accs[4] = {acc0, acc1, acc2, acc3};
#pragma unroll
  for (int t = 0; t < 4; ++t) {
    const int col = nt * 64 + t * 16 + l15;
    const float badd = bias ? bias[col] : 0.0f;
#pragma unroll
    for (int r = 0; r < 8; ++r) {
      const int row = mt * 16 + r + 8 * half;   // C/D layout: VGPR r -> M = r + 8*half
      const float v = accs[t][r] + badd;
      if (outf) outf[(size_t)row * N + col] = v;
      else      outb[(size_t)row * N + col] = f2bf(v);
    }
  }
}

// -------- flash attention, bf16 WMMA + TDM K-tile staging ---------------
// qkv layout: [B*N, 1536] = [.., q(h*64+d) | 512 + k | 1024 + v].
// Block = 128 threads (4 waves), handles 64 q-rows of one (b,h).
// K tile (32x64 bf16) is staged into LDS by the Tensor Data Mover
// (tensor_load_to_lds, TENSORcnt); V is transposed through registers.

__global__ __launch_bounds__(128) void attn_fwd_wmma(
    const u16* __restrict__ qkv, u16* __restrict__ out) {
  __shared__ __align__(16) u16 Kt[32 * 64];      // [key][d]   row-major (TDM dest)
  __shared__ __align__(16) u16 Vt[64 * 32];      // [d][key]   transposed
  __shared__ __align__(16) u16 Pt[4 * 16 * 32];  // per-wave exp(S) tile

  const int tid  = threadIdx.x;
  const int wave = tid >> 5;
  const int lane = tid & 31;
  const int l15  = lane & 15;
  const int half = lane >> 4;

  const int bid = blockIdx.x;
  const int qt  = bid & 63;
  const int h   = (bid >> 6) & 7;
  const int b   = bid >> 9;
  const int q0  = qt * 64 + wave * 16;
  const size_t RS = 1536;   // qkv row stride (bf16)

  // Q fragments live in registers for the whole kernel (Dh=64 -> 2 K-steps)
  const u16* qrow = qkv + ((size_t)(b * 4096) + q0 + l15) * RS + h * 64;
  const v16bf qa0 = load_afrag(qrow, 0, half);
  const v16bf qa1 = load_afrag(qrow, 16, half);

  float mrow[8], lrow[8];
#pragma unroll
  for (int r = 0; r < 8; ++r) { mrow[r] = -1e30f; lrow[r] = 0.0f; }
  v8f o0 = {}, o1 = {}, o2 = {}, o3 = {};
  const float scale = 0.044194173824159216f;   // 1/sqrt(512) (full dim, per ref)

  const int key = tid >> 2, seg = tid & 3;     // cooperative V loader indices
  const u16* kbase0 = qkv + (size_t)(b * 4096) * RS + 512 + h * 64;   // K, uniform
  const u16* vbase  = kbase0 + 512 + seg * 16;                        // V, per-thread

  const unsigned ldsK = (unsigned)(uintptr_t)Kt;  // LDS aperture: low 32 bits = DS addr

  for (int j0 = 0; j0 < 4096; j0 += 32) {
    __syncthreads();   // previous chunk fully consumed before overwriting LDS

    if (wave == 0) {
      // --- TDM: 2D tile load, 32 rows x 64 bf16, row stride 1536 ---------
      unsigned long long ga = (unsigned long long)(uintptr_t)(kbase0 + (size_t)j0 * RS);
      u32x4 g0;
      g0.x = 1u;                                              // count=1, user mode
      g0.y = ldsK;                                            // lds_addr
      g0.z = (unsigned)ga;                                    // global_addr[31:0]
      g0.w = ((unsigned)(ga >> 32) & 0x01FFFFFFu)             // global_addr[56:32]
             | 0x80000000u;                                   // type=2 (image)
      u32x8 g1;
      g1.s0 = 1u << 16;                                       // data_size=2B
      g1.s1 = (64u & 0xFFFFu) << 16;                          // tensor_dim0 = 64
      g1.s2 = (64u >> 16) | ((4096u & 0xFFFFu) << 16);        // tensor_dim1 = 4096
      g1.s3 = (4096u >> 16) | (64u << 16);                    // tile_dim0 = 64
      g1.s4 = 32u;                                            // tile_dim1 = 32
      g1.s5 = 1536u;                                          // tensor_dim0_stride
      g1.s6 = 0u;                                             // stride0 hi / stride1 lo
      g1.s7 = 0u;                                             // stride1 hi
      asm volatile("tensor_load_to_lds %0, %1" :: "s"(g0), "s"(g1) : "memory");
    }

    // stage V tile transposed -> Vt[d][key] (through registers)
    {
      const uint4* vp = (const uint4*)(vbase + (size_t)(j0 + key) * RS);
      union { uint4 q[2]; u16 s[16]; } uv;
      uv.q[0] = vp[0]; uv.q[1] = vp[1];
#pragma unroll
      for (int i = 0; i < 16; ++i) Vt[(seg * 16 + i) * 32 + key] = uv.s[i];
    }

    if (wave == 0) __builtin_amdgcn_s_wait_tensorcnt(0);  // K tile landed in LDS
    __syncthreads();

    // S = Q K^T for two 16-key tiles (K-dim = 64 -> 2 WMMA steps each)
    v8f s0 = {}, s1 = {};
    s0 = wmma_bf16(qa0, load_bfrag(Kt + l15 * 64,        0,  half), s0);
    s1 = wmma_bf16(qa0, load_bfrag(Kt + (16 + l15) * 64, 0,  half), s1);
    s0 = wmma_bf16(qa1, load_bfrag(Kt + l15 * 64,        16, half), s0);
    s1 = wmma_bf16(qa1, load_bfrag(Kt + (16 + l15) * 64, 16, half), s1);

    // online softmax over the 32 new columns; P -> LDS as bf16 A-tile
    u16* P = Pt + wave * 512;
#pragma unroll
    for (int r = 0; r < 8; ++r) {
      float x0 = s0[r] * scale, x1 = s1[r] * scale;
      float mx = fmaxf(x0, x1);
      mx = fmaxf(mx, __shfl_xor(mx, 1, 16));
      mx = fmaxf(mx, __shfl_xor(mx, 2, 16));
      mx = fmaxf(mx, __shfl_xor(mx, 4, 16));
      mx = fmaxf(mx, __shfl_xor(mx, 8, 16));
      float mnew  = fmaxf(mrow[r], mx);
      float alpha = __expf(mrow[r] - mnew);
      float p0 = __expf(x0 - mnew);
      float p1 = __expf(x1 - mnew);
      float ps = p0 + p1;
      ps += __shfl_xor(ps, 1, 16);
      ps += __shfl_xor(ps, 2, 16);
      ps += __shfl_xor(ps, 4, 16);
      ps += __shfl_xor(ps, 8, 16);
      lrow[r] = lrow[r] * alpha + ps;
      mrow[r] = mnew;
      o0[r] *= alpha; o1[r] *= alpha; o2[r] *= alpha; o3[r] *= alpha;
      const int prow = (r + 8 * half) * 32;     // C layout -> P row index
      P[prow + l15]      = f2bf(p0);
      P[prow + 16 + l15] = f2bf(p1);
    }
    asm volatile("" ::: "memory");  // wave-local P: DS ops are in-order per wave

    // O += P V  (P is a 16x32 A-tile, V^T rows give contiguous B-fragments)
    v16bf pa = load_afrag(P + l15 * 32, 0, half);
    o0 = wmma_bf16(pa, load_bfrag(Vt + l15 * 32,        0, half), o0);
    o1 = wmma_bf16(pa, load_bfrag(Vt + (16 + l15) * 32, 0, half), o1);
    o2 = wmma_bf16(pa, load_bfrag(Vt + (32 + l15) * 32, 0, half), o2);
    o3 = wmma_bf16(pa, load_bfrag(Vt + (48 + l15) * 32, 0, half), o3);
  }

  // normalize and write [B,N,C] bf16
  u16* obase = out + ((size_t)(b * 4096) + q0) * 512 + h * 64;
  v8f os[4] = {o0, o1, o2, o3};
#pragma unroll
  for (int r = 0; r < 8; ++r) {
    const float inv = 1.0f / lrow[r];
    const int row = r + 8 * half;
#pragma unroll
    for (int t = 0; t < 4; ++t)
      obase[(size_t)row * 512 + t * 16 + l15] = f2bf(os[t][r] * inv);
  }
}

// -------- launch ---------------------------------------------------------

extern "C" void kernel_launch(void* const* d_in, const int* in_sizes, int n_in,
                              void* d_out, int out_size, void* d_ws, size_t ws_size,
                              hipStream_t stream) {
  const float* x    = (const float*)d_in[0];   // [2,4096,512]
  const float* wqkv = (const float*)d_in[1];   // [1536,512]
  const float* wfc  = (const float*)d_in[2];   // [512,512]
  const float* bfc  = (const float*)d_in[3];   // [512]
  float* out = (float*)d_out;                  // [2,4096,512]

  char* ws = (char*)d_ws;
  u16* xb    = (u16*)(ws + 0);          // 8,388,608 B
  u16* wqkvb = (u16*)(ws + 8388608);    // 1,572,864 B
  u16* wfcb  = (u16*)(ws + 9961472);    //   524,288 B
  u16* qkvb  = (u16*)(ws + 10485760);   // 25,165,824 B
  u16* attnb = (u16*)(ws + 35651584);   // 8,388,608 B  (total ~42 MB)

  // fp32 -> bf16
  f32_to_bf16_kernel<<<4096, 256, 0, stream>>>(x,    xb,    1048576);
  f32_to_bf16_kernel<<<768,  256, 0, stream>>>(wqkv, wqkvb, 196608);
  f32_to_bf16_kernel<<<256,  256, 0, stream>>>(wfc,  wfcb,  65536);

  // qkv = x @ w_qkv^T : M=8192, N=1536, K=512 -> (512*24)/8 = 1536 blocks
  gemm_bf16_wmma<<<1536, 256, 0, stream>>>(xb, wqkvb, qkvb, nullptr, nullptr,
                                           8192, 1536, 512);

  // flash attention: B*H*(N/64) = 1024 blocks of 128 threads
  attn_fwd_wmma<<<1024, 128, 0, stream>>>(qkvb, attnb);

  // out = attn @ w_fc^T + b_fc : M=8192, N=512, K=512 -> (512*8)/8 = 512 blocks
  gemm_bf16_wmma<<<512, 256, 0, stream>>>(attnb, wfcb, nullptr, out, bfc,
                                          8192, 512, 512);
}
